// LTE_27350351741225
// MI455X (gfx1250) — matrix operand order repaired and checked by reading, not compiled, not run on hardware
//
#include <hip/hip_runtime.h>
#include <hip/hip_bf16.h>

// Problem constants (from reference)
#define BB   2
#define CC   64
#define VV   180
#define DD   384
#define WOUTN 1024
#define HIDN 256
#define PI_F 3.14159265358979323846f

typedef __attribute__((ext_vector_type(16))) __bf16 v16bf;
typedef __attribute__((ext_vector_type(8)))  float  v8f;

union Frag { v16bf v; unsigned int u[8]; unsigned short s[16]; };

__device__ __forceinline__ unsigned short f2bf(float x) {
    unsigned int u = __float_as_uint(x);
    u += 0x7FFFu + ((u >> 16) & 1u);          // round-to-nearest-even
    return (unsigned short)(u >> 16);
}
__device__ __forceinline__ float bf2f(unsigned short s) {
    return __uint_as_float(((unsigned int)s) << 16);
}

// Scheduling-hint masks (LLVM AMDGPU sched_group_barrier)
#define SG_WMMA   0x008   // MFMA/WMMA
#define SG_DSREAD 0x100   // DS read

// ---------------------------------------------------------------------------
// Zero output (harness poisons d_out)
// ---------------------------------------------------------------------------
__global__ void lte_zero(float* __restrict__ out) {
    int t = blockIdx.x * 256 + threadIdx.x;
    if (t < BB * WOUTN) out[t] = 0.f;
}

// ---------------------------------------------------------------------------
// Backprojection of middle channel: bilinear sample + weighted sum over views
// ---------------------------------------------------------------------------
__global__ void lte_res(const float* __restrict__ sino, const float* __restrict__ grid,
                        const float* __restrict__ sqinv, float* __restrict__ out) {
    int t = blockIdx.x * 256 + threadIdx.x;
    if (t >= BB * WOUTN) return;
    int b = t >> 10;
    int q = t & (WOUTN - 1);
    const float* img = sino + ((size_t)(b * CC + CC / 2) * VV) * DD;
    float sum = 0.f;
    for (int v = 0; v < VV; ++v) {
        int gi = ((b * VV + v) * WOUTN + q) * 2;
        float gx = grid[gi], gy = grid[gi + 1];
        float ix = fminf(fmaxf(((gx + 1.f) * DD - 1.f) * 0.5f, 0.f), (float)(DD - 1));
        float iy = fminf(fmaxf(((gy + 1.f) * VV - 1.f) * 0.5f, 0.f), (float)(VV - 1));
        float x0f = floorf(ix), y0f = floorf(iy);
        float wx = ix - x0f, wy = iy - y0f;
        int x0 = (int)x0f, y0 = (int)y0f;
        int x1 = min(x0 + 1, DD - 1), y1 = min(y0 + 1, VV - 1);
        float v00 = img[y0 * DD + x0], v01 = img[y0 * DD + x1];
        float v10 = img[y1 * DD + x0], v11 = img[y1 * DD + x1];
        float val = (v00 * (1.f - wx) + v01 * wx) * (1.f - wy)
                  + (v10 * (1.f - wx) + v11 * wx) * wy;
        sum += val * sqinv[(b * VV + v) * WOUTN + q] * 10000.f;
    }
    atomicAdd(&out[t], sum);
}

// ---------------------------------------------------------------------------
// Dual-head 3x3 conv (64->64) as WMMA im2col GEMM, K = 9*64 = 576.
// WG = (b, row y, 64-wide x block); 8 waves = 4 M-tiles x 2 heads; the staged
// input tile in LDS is shared by both heads. Software-pipelined: next kt's
// A/B fragments (12 DS reads) are issued before this kt's 4 WMMAs, and
// sched_group_barrier pins the [DS*12][WMMA*4] batching so waits are partial.
// ---------------------------------------------------------------------------
#define CONV_IN_BYTES   26144                  // 3*66*66 halves, rounded to 32
#define CONV_WF_BYTES   147456                 // 2*18*4*512 halves
#define CONV_LDS_BYTES  (CONV_IN_BYTES + CONV_WF_BYTES + 512)

__device__ __forceinline__ void conv_loadA(Frag& A, const unsigned short* inT,
                                           int kt, int mt, int nl, int h) {
    int kxy = kt >> 1, icb = (kt & 1) * 32;
    int ky = kxy / 3, kx = kxy % 3;
    int colb = mt * 16 + nl + kx;
    const unsigned short* base = &inT[(ky * 66 + colb) * 66 + icb];
    #pragma unroll
    for (int p = 0; p < 8; ++p) {
        int kl = 2 * p + 16 * h;
        A.u[p] = *(const unsigned int*)&base[kl];
    }
}
__device__ __forceinline__ void conv_loadB(Frag* Bf, const unsigned short* Wf,
                                           int head, int kt, int lane) {
    #pragma unroll
    for (int nt = 0; nt < 4; ++nt) {
        const unsigned short* wp = &Wf[(((head * 18 + kt) * 4 + nt) << 9) + lane * 16];
        *(uint4*)&Bf[nt].u[0] = *(const uint4*)wp;
        *(uint4*)&Bf[nt].u[4] = *(const uint4*)(wp + 8);
    }
}

__global__ void lte_conv(const float* __restrict__ sino,
                         const float* __restrict__ coef_w, const float* __restrict__ coef_b,
                         const float* __restrict__ freq_w, const float* __restrict__ freq_b,
                         unsigned short* __restrict__ coefWS, float* __restrict__ freqWS) {
    extern __shared__ char smem[];
    unsigned short* inT   = (unsigned short*)smem;
    unsigned short* Wf    = (unsigned short*)(smem + CONV_IN_BYTES);
    float*          biasL = (float*)(smem + CONV_IN_BYTES + CONV_WF_BYTES);

    int tid = threadIdx.x;
    int blk = blockIdx.x;
    int xb = blk % 6;
    int y  = (blk / 6) % VV;
    int b  = blk / (6 * VV);
    int x0 = xb * 64;

    // Stage input halo tile -> LDS [ky][col][ic(+2 pad)] bf16.
    // Loop order: col fastest => coalesced global reads along detector axis.
    for (int f = tid; f < 3 * 64 * 66; f += 256) {
        int col = f % 66;
        int t2  = f / 66;
        int ic  = t2 & 63;
        int ky  = t2 >> 6;
        int gy = y + ky - 1;
        int gx = x0 + col - 1;
        float v = 0.f;
        if (gy >= 0 && gy < VV && gx >= 0 && gx < DD)
            v = sino[((b * CC + ic) * VV + gy) * DD + gx];
        inT[(ky * 66 + col) * 66 + ic] = f2bf(v);
    }
    // Stage weights (both heads): linear coalesced global read, scatter into
    // B-fragment layout. k ordering: k = (ky*3+kx)*64 + ic.
    for (int g = tid; g < 2 * 36864; g += 256) {
        int head = g / 36864;
        int r    = g % 36864;                  // flat [oc][ic][ky][kx]
        int oc   = r / 576;
        int r2   = r % 576;
        int ic   = r2 / 9;
        int kxy  = r2 % 9;
        int k    = kxy * 64 + ic;
        int kt   = k >> 5;
        int kin  = k & 31;
        int lane_ = ((kin >> 4) << 4) | (oc & 15);
        int j     = kin & 15;
        int nt    = oc >> 4;
        const float* gw = head ? freq_w : coef_w;
        Wf[(((head * 18 + kt) * 4 + nt) << 9) + lane_ * 16 + j] = f2bf(gw[r]);
    }
    if (tid < 128) biasL[tid] = (tid < 64) ? coef_b[tid] : freq_b[tid - 64];
    __syncthreads();

    int w    = tid >> 5;
    int lane = tid & 31;
    int head = w >> 2;
    int mt   = w & 3;
    int h    = lane >> 4;
    int nl   = lane & 15;

    v8f acc[4];
    #pragma unroll
    for (int nt = 0; nt < 4; ++nt) {
        float bias = biasL[head * 64 + nt * 16 + nl];
        #pragma unroll
        for (int r = 0; r < 8; ++r) acc[nt][r] = bias;
    }

    // Software-pipelined K loop with ping-pong fragment buffers.
    Frag Ad[2];
    Frag Bd[2][4];
    conv_loadA(Ad[0], inT, 0, mt, nl, h);
    conv_loadB(Bd[0], Wf, head, 0, lane);
    #pragma unroll
    for (int kt = 0; kt < 18; ++kt) {
        int cur = kt & 1, nxt = cur ^ 1;
        if (kt < 17) {
            conv_loadA(Ad[nxt], inT, kt + 1, mt, nl, h);
            conv_loadB(Bd[nxt], Wf, head, kt + 1, lane);
        }
        #pragma unroll
        for (int nt = 0; nt < 4; ++nt) {
            acc[nt] = __builtin_amdgcn_wmma_f32_16x16x32_bf16(
                false, Ad[cur].v, false, Bd[cur][nt].v, (short)0, acc[nt], false, false);
        }
        __builtin_amdgcn_sched_group_barrier(SG_DSREAD, 12, 0);
        __builtin_amdgcn_sched_group_barrier(SG_WMMA,    4, 0);
    }

    #pragma unroll
    for (int nt = 0; nt < 4; ++nt) {
        int oc = nt * 16 + nl;
        #pragma unroll
        for (int r = 0; r < 8; ++r) {
            int px  = x0 + mt * 16 + r + 8 * h;      // C layout: row = r + 8*half
            int idx = ((b * CC + oc) * VV + y) * DD + px;
            if (head == 0) coefWS[idx] = f2bf(acc[nt][r]);
            else           freqWS[idx] = acc[nt][r];
        }
    }
}

// ---------------------------------------------------------------------------
// Fused query kernel: nearest-gather + LTE feature build + 64->256->256->1 MLP
// via WMMA bf16, blend over the two detector shifts, atomic reduce over views.
// WG: 64 queries x 2 shifts = 128 rows; wave m-tile = 16 rows.
// ---------------------------------------------------------------------------
#define Q_W0F_OFF   0         // 16384 halves  (32768 B)
#define Q_W1F_OFF   32768     // 65536 halves  (131072 B)
#define Q_B0_OFF    163840    // 256 f32
#define Q_B1_OFF    164864    // 256 f32
#define Q_W2_OFF    165888    // 256 f32
#define Q_XB_OFF    166912    // 8 waves * 16*66 halves (16896 B)
#define Q_HB_OFF    183808    // 8 waves * 16*264 halves (67584 B)
#define Q_PRED_OFF  251392    // 128 f32
#define Q_LEN_OFF   251904    // 128 f32
#define Q_LDS_BYTES 252416

__device__ __forceinline__ void mlp_loadB4(Frag* Bf, const unsigned short* Wfrag,
                                           int nt, int g4, int lane) {
    #pragma unroll
    for (int q4 = 0; q4 < 4; ++q4) {
        int kt = g4 * 4 + q4;
        const unsigned short* wp = &Wfrag[((kt * 16 + nt) << 9) + lane * 16];
        *(uint4*)&Bf[q4].u[0] = *(const uint4*)wp;
        *(uint4*)&Bf[q4].u[4] = *(const uint4*)(wp + 8);
    }
}

__global__ void lte_query(const float* __restrict__ grid, const float* __restrict__ sqinv,
                          const int* __restrict__ scale_p, const float* __restrict__ phw,
                          const float* __restrict__ w0, const float* __restrict__ b0,
                          const float* __restrict__ w1, const float* __restrict__ b1,
                          const float* __restrict__ w2, const float* __restrict__ b2,
                          const unsigned short* __restrict__ coefWS,
                          const float* __restrict__ freqWS,
                          float* __restrict__ out) {
    extern __shared__ char smem[];
    unsigned short* W0f  = (unsigned short*)(smem + Q_W0F_OFF);
    unsigned short* W1f  = (unsigned short*)(smem + Q_W1F_OFF);
    float* b0s  = (float*)(smem + Q_B0_OFF);
    float* b1s  = (float*)(smem + Q_B1_OFF);
    float* w2s  = (float*)(smem + Q_W2_OFF);
    unsigned short* xbuf = (unsigned short*)(smem + Q_XB_OFF);
    unsigned short* hbuf = (unsigned short*)(smem + Q_HB_OFF);
    float* predb = (float*)(smem + Q_PRED_OFF);
    float* lenb  = (float*)(smem + Q_LEN_OFF);

    int tid = threadIdx.x;

    // Stage W0 [64,256] / W1 [256,256]: linear coalesced global reads,
    // scatter into bf16 B-fragment layout.
    for (int g = tid; g < 64 * HIDN; g += 256) {
        int k = g >> 8, n = g & 255;
        int kt = k >> 5, kin = k & 31;
        int lane_ = ((kin >> 4) << 4) | (n & 15);
        int j = kin & 15;
        int nt = n >> 4;
        W0f[((kt * 16 + nt) << 9) + lane_ * 16 + j] = f2bf(w0[g]);
    }
    for (int g = tid; g < HIDN * HIDN; g += 256) {
        int k = g >> 8, n = g & 255;
        int kt = k >> 5, kin = k & 31;
        int lane_ = ((kin >> 4) << 4) | (n & 15);
        int j = kin & 15;
        int nt = n >> 4;
        W1f[((kt * 16 + nt) << 9) + lane_ * 16 + j] = f2bf(w1[g]);
    }
    { b0s[tid] = b0[tid]; b1s[tid] = b1[tid]; w2s[tid] = w2[tid]; }
    __syncthreads();

    int w    = tid >> 5;
    int lane = tid & 31;
    int h    = lane >> 4;
    int nl   = lane & 15;
    int qbase = blockIdx.x * 64;
    float cell = 2.f / (float)scale_p[0];
    const float S = 1.f / (float)DD + 1e-6f;

    unsigned short* xb = xbuf + w * 1056;   // [16][66] bf16, row-major
    unsigned short* hb = hbuf + w * 4224;   // [16][264] bf16, row-major

    // ---- Phase B: build x = q_coef * concat(cos(pi fr), sin(pi fr)) --------
    for (int r = 0; r < 16; ++r) {
        int ridx = w * 16 + r;
        int qi   = qbase + (ridx >> 1);     // rows interleave shift0/shift1
        int sh   = ridx & 1;
        int b    = qi / (VV * WOUTN);
        float gx = grid[qi * 2 + 0];
        float gy = grid[qi * 2 + 1];
        float gxs = gx + (sh ? S : -S);
        float ixf = fminf(fmaxf(((gxs + 1.f) * DD - 1.f) * 0.5f, 0.f), (float)(DD - 1));
        float iyf = fminf(fmaxf(((gy  + 1.f) * VV - 1.f) * 0.5f, 0.f), (float)(VV - 1));
        int xi = (int)rintf(ixf); xi = min(max(xi, 0), DD - 1);
        int yi = (int)rintf(iyf); yi = min(max(yi, 0), VV - 1);
        float qcoord = (2.f * (float)xi + 1.f) / (float)DD - 1.f;
        float rel = (gx - qcoord) * (float)DD;   // uses UNshifted grid x
        if (lane == 0) lenb[ridx] = fabsf(rel) + 1e-4f;
        int base = (b * CC) * (VV * DD) + yi * DD + xi;
        #pragma unroll
        for (int it = 0; it < 2; ++it) {
            int c = lane + it * 32;
            int j = c & 31;
            float f2 = freqWS[base + (2 * j)     * (VV * DD)]
                     + freqWS[base + (2 * j + 1) * (VV * DD)];
            float fr = rel * f2 + cell * phw[j];
            float sv, cv;
            __sincosf(PI_F * fr, &sv, &cv);
            float qv = (c < 32) ? cv : sv;
            float cf = bf2f(coefWS[base + c * (VV * DD)]);
            xb[r * 66 + c] = f2bf(cf * qv);
        }
    }
    __syncthreads();

    // ---- GEMM1: [16,64] x [64,256] + b0, ReLU -> hbuf ----------------------
    {
        Frag A[2];
        #pragma unroll
        for (int kt = 0; kt < 2; ++kt) {
            #pragma unroll
            for (int p = 0; p < 8; ++p) {
                int kl = 2 * p + 16 * h;
                A[kt].u[p] = *(const unsigned int*)&xb[nl * 66 + kt * 32 + kl];
            }
        }
        for (int nt = 0; nt < 16; ++nt) {
            float bias = b0s[nt * 16 + nl];
            v8f acc;
            #pragma unroll
            for (int r = 0; r < 8; ++r) acc[r] = bias;
            Frag Bf[2];
            #pragma unroll
            for (int kt = 0; kt < 2; ++kt) {
                const unsigned short* wp = &W0f[((kt * 16 + nt) << 9) + lane * 16];
                *(uint4*)&Bf[kt].u[0] = *(const uint4*)wp;
                *(uint4*)&Bf[kt].u[4] = *(const uint4*)(wp + 8);
            }
            #pragma unroll
            for (int kt = 0; kt < 2; ++kt) {
                acc = __builtin_amdgcn_wmma_f32_16x16x32_bf16(
                    false, A[kt].v, false, Bf[kt].v, (short)0, acc, false, false);
            }
            __builtin_amdgcn_sched_group_barrier(SG_DSREAD, 4, 0);
            __builtin_amdgcn_sched_group_barrier(SG_WMMA,   2, 0);
            #pragma unroll
            for (int r = 0; r < 8; ++r) {
                float vv = fmaxf(acc[r], 0.f);
                hb[(r + 8 * h) * 264 + nt * 16 + nl] = f2bf(vv);
            }
        }
    }
    __syncthreads();

    // ---- GEMM2: [16,256] x [256,256] + b1, ReLU; GEMM3 folded as dot(w2) ---
    {
        Frag A[8];
        #pragma unroll
        for (int kt = 0; kt < 8; ++kt) {
            #pragma unroll
            for (int p = 0; p < 8; ++p) {
                int kl = 2 * p + 16 * h;
                A[kt].u[p] = *(const unsigned int*)&hb[nl * 264 + kt * 32 + kl];
            }
        }
        float psum[8];
        #pragma unroll
        for (int r = 0; r < 8; ++r) psum[r] = 0.f;

        // Double-buffered B fragments: group0 loaded ahead of the iteration,
        // group1 loaded at iteration start, next nt's group0 loaded between
        // the two WMMA quads.
        Frag Bf[4], Bn[4];
        mlp_loadB4(Bf, W1f, 0, 0, lane);
        for (int nt = 0; nt < 16; ++nt) {
            mlp_loadB4(Bn, W1f, nt, 1, lane);
            float bias = b1s[nt * 16 + nl];
            v8f acc;
            #pragma unroll
            for (int r = 0; r < 8; ++r) acc[r] = bias;
            #pragma unroll
            for (int q4 = 0; q4 < 4; ++q4) {
                acc = __builtin_amdgcn_wmma_f32_16x16x32_bf16(
                    false, A[q4].v, false, Bf[q4].v, (short)0, acc, false, false);
            }
            int ntn = (nt + 1) & 15;
            mlp_loadB4(Bf, W1f, ntn, 0, lane);
            #pragma unroll
            for (int q4 = 0; q4 < 4; ++q4) {
                acc = __builtin_amdgcn_wmma_f32_16x16x32_bf16(
                    false, A[4 + q4].v, false, Bn[q4].v, (short)0, acc, false, false);
            }
            float wv = w2s[nt * 16 + nl];
            #pragma unroll
            for (int r = 0; r < 8; ++r) psum[r] += fmaxf(acc[r], 0.f) * wv;
            __builtin_amdgcn_sched_group_barrier(SG_DSREAD, 8, 0);
            __builtin_amdgcn_sched_group_barrier(SG_WMMA,   4, 0);
            __builtin_amdgcn_sched_group_barrier(SG_DSREAD, 8, 0);
            __builtin_amdgcn_sched_group_barrier(SG_WMMA,   4, 0);
        }
        // reduce partials across the 16 lanes of each half (wave32 shuffles)
        #pragma unroll
        for (int off = 1; off < 16; off <<= 1) {
            #pragma unroll
            for (int r = 0; r < 8; ++r) psum[r] += __shfl_xor(psum[r], off, 32);
        }
        if (nl == 0) {
            float b2v = b2[0];
            #pragma unroll
            for (int r = 0; r < 8; ++r) predb[w * 16 + 8 * h + r] = psum[r] + b2v;
        }
    }
    __syncthreads();

    // ---- Blend shifts (swapped length weighting) and reduce over views -----
    if (tid < 64) {
        int qi = qbase + tid;
        int b  = qi / (VV * WOUTN);
        int qq = qi % WOUTN;
        float p0 = predb[2 * tid],     p1 = predb[2 * tid + 1];
        float l0 = lenb[2 * tid],      l1 = lenb[2 * tid + 1];
        float ret = (p0 * l1 + p1 * l0) / (l0 + l1);
        float si  = sqinv[qi];
        atomicAdd(&out[b * WOUTN + qq], ret * si * 10000.f);
    }
}

// ---------------------------------------------------------------------------
extern "C" void kernel_launch(void* const* d_in, const int* in_sizes, int n_in,
                              void* d_out, int out_size, void* d_ws, size_t ws_size,
                              hipStream_t stream) {
    (void)in_sizes; (void)n_in; (void)out_size; (void)ws_size;
    const float* sino   = (const float*)d_in[0];
    const float* grid   = (const float*)d_in[1];
    const float* sqinv  = (const float*)d_in[2];
    const int*   scale  = (const int*)  d_in[3];
    const float* coef_w = (const float*)d_in[4];
    const float* coef_b = (const float*)d_in[5];
    const float* freq_w = (const float*)d_in[6];
    const float* freq_b = (const float*)d_in[7];
    const float* phw    = (const float*)d_in[8];
    const float* w0     = (const float*)d_in[9];
    const float* b0     = (const float*)d_in[10];
    const float* w1     = (const float*)d_in[11];
    const float* b1     = (const float*)d_in[12];
    const float* w2     = (const float*)d_in[13];
    const float* b2     = (const float*)d_in[14];
    float* out = (float*)d_out;

    char* ws = (char*)d_ws;
    unsigned short* coefWS = (unsigned short*)ws;                                  // bf16 [B,64,V,D]
    float*          freqWS = (float*)(ws + (size_t)BB * CC * VV * DD * 2);         // f32  [B,64,V,D]

    lte_zero<<<8, 256, 0, stream>>>(out);
    lte_conv<<<BB * VV * (DD / 64), 256, CONV_LDS_BYTES, stream>>>(
        sino, coef_w, coef_b, freq_w, freq_b, coefWS, freqWS);
    lte_res<<<8, 256, 0, stream>>>(sino, grid, sqinv, out);
    lte_query<<<BB * VV * WOUTN / 64, 256, Q_LDS_BYTES, stream>>>(
        grid, sqinv, scale, phw, w0, b0, w1, b1, w2, b2, coefWS, freqWS, out);
}
